// PlacementNetwork_90022514524579
// MI455X (gfx1250) — compile-verified, matrix operand order checked by reading, not run on hardware
//
#include <hip/hip_runtime.h>
#include <hip/hip_bf16.h>
#include <stdint.h>

// ---------------------------------------------------------------------------
// Problem constants (from reference)
// ---------------------------------------------------------------------------
constexpr int Bn   = 2;        // batch
constexpr int Nn   = 50000;    // nodes
constexpr int En   = 800000;   // edges
constexpr int Fn   = 16;       // node feature dim
constexpr int EMB  = 32;       // embedding dim
constexpr int Ln   = 3;        // GNN layers
constexpr int Rr   = 100, Cc = 100;
constexpr float NEGV = -100000000.0f;

typedef __attribute__((ext_vector_type(16))) _Float16 v16h;
typedef __attribute__((ext_vector_type(8)))  _Float16 v8h;
typedef __attribute__((ext_vector_type(8)))  float    v8f;

// ---------------------------------------------------------------------------
// Generic zero
// ---------------------------------------------------------------------------
__global__ void zero_f32_kernel(float* p, long n) {
  long i = (long)blockIdx.x * blockDim.x + threadIdx.x;
  long s = (long)gridDim.x * blockDim.x;
  for (; i < n; i += s) p[i] = 0.0f;
}

// ---------------------------------------------------------------------------
// Pack msg_w / upd_w f32 -> f16 WMMA B-fragment lane layout.
// B frag (K=32, N=16): lane = {kh:1, n:4}; element e in v16h:
//   e<8  -> K = kh*8 + e
//   e>=8 -> K = kh*8 + 16 + (e-8)
// frag index: [l][kstep][ntile][lane][e]
// ---------------------------------------------------------------------------
__global__ void pack_weights_kernel(const float* __restrict__ msg_w,
                                    const float* __restrict__ upd_w,
                                    _Float16* __restrict__ pw_msg,
                                    _Float16* __restrict__ pw_upd) {
  const int per = Ln * 2 * 2 * 512;   // 6144 halves per table
  int stride = gridDim.x * blockDim.x;
  for (int i = blockIdx.x * blockDim.x + threadIdx.x; i < 2 * per; i += stride) {
    int which = i / per;              // 0 = msg, 1 = upd
    int r     = i % per;
    int l    = r / (2 * 2 * 512);
    int rem  = r % (2 * 2 * 512);
    int s    = rem / (2 * 512);
    int rem2 = rem % (2 * 512);
    int nt   = rem2 / 512;
    int li   = rem2 % 512;
    int lane = li / 16, e = li % 16;
    int n  = lane & 15, kh = lane >> 4;
    int kk = (e < 8) ? (kh * 8 + e) : (kh * 8 + 16 + (e - 8));
    int k  = s * 32 + kk;
    int col = nt * 16 + n;
    float v = (which == 0) ? msg_w[(l * 65 + k) * EMB + col]
                           : upd_w[(l * 64 + k) * EMB + col];
    (which == 0 ? pw_msg : pw_upd)[r] = (_Float16)v;
  }
}

// ---------------------------------------------------------------------------
// Node projection: x16 = f16(node_features @ W + b)   (B,N,16)@(16,32)
// ---------------------------------------------------------------------------
__global__ void __launch_bounds__(256)
node_proj_kernel(const float* __restrict__ nf, const float* __restrict__ w,
                 const float* __restrict__ bias, _Float16* __restrict__ x16) {
  __shared__ float sw[Fn * EMB];
  __shared__ float sb[EMB];
  for (int i = threadIdx.x; i < Fn * EMB; i += blockDim.x) sw[i] = w[i];
  if (threadIdx.x < EMB) sb[threadIdx.x] = bias[threadIdx.x];
  __syncthreads();
  long total = (long)Bn * Nn;
  long stride = (long)gridDim.x * blockDim.x;
  for (long idx = (long)blockIdx.x * blockDim.x + threadIdx.x; idx < total; idx += stride) {
    const float* f = nf + idx * Fn;
    float fr[Fn];
#pragma unroll
    for (int k = 0; k < Fn; k++) fr[k] = f[k];
    _Float16* xo = x16 + idx * EMB;
#pragma unroll
    for (int c = 0; c < EMB; c++) {
      float acc = sb[c];
#pragma unroll
      for (int k = 0; k < Fn; k++) acc += fr[k] * sw[k * EMB + c];
      xo[c] = (_Float16)acc;
    }
  }
}

// ---------------------------------------------------------------------------
// Degree (dst-indexed) and reciprocal
// ---------------------------------------------------------------------------
__global__ void degree_kernel(const int* __restrict__ dstIdx, float* __restrict__ deg) {
  int stride = gridDim.x * blockDim.x;
  for (int e = blockIdx.x * blockDim.x + threadIdx.x; e < En; e += stride)
    atomicAdd(&deg[dstIdx[e]], 1.0f);
}
__global__ void invdeg_kernel(float* deg) {
  int stride = gridDim.x * blockDim.x;
  for (int n = blockIdx.x * blockDim.x + threadIdx.x; n < Nn; n += stride)
    deg[n] = 1.0f / fmaxf(deg[n], 1.0f);
}

// ---------------------------------------------------------------------------
// Edge-message MLP + fused scatter-add aggregation.
// One wave handles 16 edges per tile:
//   msgs = relu([x_src | x_dst] @ W[0:64] + ew*W[64] + b)
//   agg[dst] += msgs * invdeg[dst]
// 4x v_wmma_f32_16x16x32_f16 per tile (2 K-steps x 2 N-tiles).
// ---------------------------------------------------------------------------
__global__ void __launch_bounds__(256)
msg_kernel(const _Float16* __restrict__ x16, const int* __restrict__ eidx,
           const float* __restrict__ ew, const float* __restrict__ invdeg,
           const _Float16* __restrict__ pw, const float* __restrict__ msg_w_raw,
           const float* __restrict__ msg_b, float* __restrict__ agg, int l) {
  const int lane   = threadIdx.x & 31;
  const int waveId = blockIdx.x * (blockDim.x >> 5) + (threadIdx.x >> 5);
  const int nWaves = gridDim.x * (blockDim.x >> 5);
  const int m  = lane & 15;
  const int kh = lane >> 4;

  const _Float16* pb = pw + (size_t)l * 2 * 2 * 512;
  v16h b00 = *(const v16h*)(pb + (0 * 2 + 0) * 512 + lane * 16);
  v16h b01 = *(const v16h*)(pb + (0 * 2 + 1) * 512 + lane * 16);
  v16h b10 = *(const v16h*)(pb + (1 * 2 + 0) * 512 + lane * 16);
  v16h b11 = *(const v16h*)(pb + (1 * 2 + 1) * 512 + lane * 16);
  const float w64a = msg_w_raw[(l * 65 + 64) * EMB + m];
  const float w64b = msg_w_raw[(l * 65 + 64) * EMB + 16 + m];
  const float ba = msg_b[l * EMB + m];
  const float bb = msg_b[l * EMB + 16 + m];

  const int tilesPerB = En / 16;
  const int total = Bn * tilesPerB;
  const int* srcIdx = eidx;
  const int* dstIdx = eidx + En;

  for (int t = waveId; t < total; t += nWaves) {
    int b    = t / tilesPerB;
    int tile = t - b * tilesPerB;
    int e0   = tile * 16;
    int eg   = e0 + m;
    // speculative prefetch of next tile's indices (dropped if OOB)
    __builtin_prefetch(srcIdx + eg + (size_t)nWaves * 16, 0, 1);
    int sn = srcIdx[eg];
    int dn = dstIdx[eg];
    const _Float16* ps = x16 + ((size_t)b * Nn + sn) * EMB + kh * 8;
    const _Float16* pd = x16 + ((size_t)b * Nn + dn) * EMB + kh * 8;
    v8h s_lo = *(const v8h*)ps, s_hi = *(const v8h*)(ps + 16);
    v8h d_lo = *(const v8h*)pd, d_hi = *(const v8h*)(pd + 16);
    v16h a0, a1;
#pragma unroll
    for (int i = 0; i < 8; i++) {
      a0[i] = s_lo[i]; a0[i + 8] = s_hi[i];
      a1[i] = d_lo[i]; a1[i + 8] = d_hi[i];
    }
    v8f c0 = {}; v8f c1 = {};
    c0 = __builtin_amdgcn_wmma_f32_16x16x32_f16(false, a0, false, b00, (short)0, c0, false, false);
    c1 = __builtin_amdgcn_wmma_f32_16x16x32_f16(false, a0, false, b01, (short)0, c1, false, false);
    c0 = __builtin_amdgcn_wmma_f32_16x16x32_f16(false, a1, false, b10, (short)0, c0, false, false);
    c1 = __builtin_amdgcn_wmma_f32_16x16x32_f16(false, a1, false, b11, (short)0, c1, false, false);
    // epilogue: +ew rank-1, +bias, relu, *invdeg[dst], scatter-add
#pragma unroll
    for (int j = 0; j < 8; j++) {
      int row = kh * 8 + j;          // C/D layout: VGPR j -> M = j + 8*kh, N = lane&15
      int er  = e0 + row;
      float wgt = ew[er];
      int   dr  = dstIdx[er];
      float sc  = invdeg[dr];
      float v0 = fmaxf(c0[j] + wgt * w64a + ba, 0.0f) * sc;
      float v1 = fmaxf(c1[j] + wgt * w64b + bb, 0.0f) * sc;
      float* pagg = agg + ((size_t)b * Nn + dr) * EMB;
      atomicAdd(pagg + m,      v0);
      atomicAdd(pagg + 16 + m, v1);
    }
  }
}

// ---------------------------------------------------------------------------
// Node update: x_out = relu([x | agg] @ upd_w + b), one wave per 16-node tile.
// ---------------------------------------------------------------------------
__global__ void __launch_bounds__(256)
upd_kernel(const _Float16* __restrict__ xin, const float* __restrict__ agg,
           const _Float16* __restrict__ pw, const float* __restrict__ upd_b,
           _Float16* __restrict__ xout, int l) {
  const int lane   = threadIdx.x & 31;
  const int waveId = blockIdx.x * (blockDim.x >> 5) + (threadIdx.x >> 5);
  const int nWaves = gridDim.x * (blockDim.x >> 5);
  const int m  = lane & 15;
  const int kh = lane >> 4;

  const _Float16* pb = pw + (size_t)l * 2 * 2 * 512;
  v16h b00 = *(const v16h*)(pb + (0 * 2 + 0) * 512 + lane * 16);
  v16h b01 = *(const v16h*)(pb + (0 * 2 + 1) * 512 + lane * 16);
  v16h b10 = *(const v16h*)(pb + (1 * 2 + 0) * 512 + lane * 16);
  v16h b11 = *(const v16h*)(pb + (1 * 2 + 1) * 512 + lane * 16);
  const float ba = upd_b[l * EMB + m];
  const float bb = upd_b[l * EMB + 16 + m];

  const int tilesPerB = Nn / 16;     // 3125
  const int total = Bn * tilesPerB;

  for (int t = waveId; t < total; t += nWaves) {
    int b    = t / tilesPerB;
    int tile = t - b * tilesPerB;
    int n0   = tile * 16;
    int n    = n0 + m;
    const _Float16* px = xin + ((size_t)b * Nn + n) * EMB + kh * 8;
    const float*    pa = agg + ((size_t)b * Nn + n) * EMB + kh * 8;
    v8h xlo = *(const v8h*)px, xhi = *(const v8h*)(px + 16);
    v16h a0, a1;
#pragma unroll
    for (int i = 0; i < 8; i++) {
      a0[i] = xlo[i];             a0[i + 8] = xhi[i];
      a1[i] = (_Float16)pa[i];    a1[i + 8] = (_Float16)pa[i + 16];
    }
    v8f c0 = {}; v8f c1 = {};
    c0 = __builtin_amdgcn_wmma_f32_16x16x32_f16(false, a0, false, b00, (short)0, c0, false, false);
    c1 = __builtin_amdgcn_wmma_f32_16x16x32_f16(false, a0, false, b01, (short)0, c1, false, false);
    c0 = __builtin_amdgcn_wmma_f32_16x16x32_f16(false, a1, false, b10, (short)0, c0, false, false);
    c1 = __builtin_amdgcn_wmma_f32_16x16x32_f16(false, a1, false, b11, (short)0, c1, false, false);
#pragma unroll
    for (int j = 0; j < 8; j++) {
      int row  = kh * 8 + j;
      int node = n0 + row;
      _Float16* po = xout + ((size_t)b * Nn + node) * EMB;
      po[m]      = (_Float16)fmaxf(c0[j] + ba, 0.0f);
      po[16 + m] = (_Float16)fmaxf(c1[j] + bb, 0.0f);
    }
  }
}

// ---------------------------------------------------------------------------
// Graph mean: per-(b,c) block reduction over N nodes.
// ---------------------------------------------------------------------------
__global__ void __launch_bounds__(256)
gsum_kernel(const _Float16* __restrict__ x, float* __restrict__ gsum) {
  int b = blockIdx.x / EMB, c = blockIdx.x % EMB;
  float acc = 0.0f;
  for (int n = threadIdx.x; n < Nn; n += blockDim.x)
    acc += (float)x[((size_t)b * Nn + n) * EMB + c];
  __shared__ float red[256];
  red[threadIdx.x] = acc;
  __syncthreads();
  for (int s = 128; s > 0; s >>= 1) {
    if (threadIdx.x < s) red[threadIdx.x] += red[threadIdx.x + s];
    __syncthreads();
  }
  if (threadIdx.x == 0) gsum[blockIdx.x] = red[0];
}

// ---------------------------------------------------------------------------
// Heads: combined = [graph_mean | macro | meta], policy pre-deconv, value.
// Tiny — single block.
// ---------------------------------------------------------------------------
__global__ void __launch_bounds__(256)
head_kernel(const _Float16* __restrict__ xfin, const float* __restrict__ gsum,
            const int* __restrict__ macro_idx, const float* __restrict__ metadata,
            const float* __restrict__ macro_w, const float* __restrict__ macro_b,
            const float* __restrict__ meta_w, const float* __restrict__ meta_b,
            const float* __restrict__ policy_w, const float* __restrict__ policy_b,
            const float* __restrict__ val_w1, const float* __restrict__ val_b1,
            const float* __restrict__ val_w2, const float* __restrict__ val_b2,
            float* __restrict__ pol, float* __restrict__ value_out) {
  __shared__ float comb[Bn][80];
  __shared__ float vh[Bn][EMB];
  int t = threadIdx.x;
  if (t < Bn * 80) {
    int b = t / 80, j = t % 80;
    float v;
    if (j < 32) {
      v = gsum[b * EMB + j] * (1.0f / (float)Nn);
    } else if (j < 64) {
      int c = j - 32;
      int mi = macro_idx[b];
      float acc = macro_b[c];
      const _Float16* xr = xfin + ((size_t)b * Nn + mi) * EMB;
      for (int k = 0; k < EMB; k++) acc += (float)xr[k] * macro_w[k * EMB + c];
      v = acc;
    } else {
      int c = j - 64;
      float acc = meta_b[c];
      for (int k = 0; k < 4; k++) acc += metadata[b * 4 + k] * meta_w[k * 16 + c];
      v = fmaxf(acc, 0.0f);
    }
    comb[b][j] = v;
  }
  __syncthreads();
  for (int o = t; o < Bn * 512; o += blockDim.x) {
    int b = o / 512, u = o % 512;
    float acc = policy_b[u];
    for (int k = 0; k < 80; k++) acc += comb[b][k] * policy_w[k * 512 + u];
    pol[o] = fmaxf(acc, 0.0f);
  }
  if (t < Bn * EMB) {
    int b = t / EMB, u = t % EMB;
    float acc = val_b1[u];
    for (int k = 0; k < 80; k++) acc += comb[b][k] * val_w1[k * EMB + u];
    vh[b][u] = fmaxf(acc, 0.0f);
  }
  __syncthreads();
  if (t < Bn) {
    float acc = val_b2[0];
    for (int k = 0; k < EMB; k++) acc += vh[t][k] * val_w2[k];
    value_out[t] = acc;
  }
}

// ---------------------------------------------------------------------------
// ConvTranspose2d(k=4, s=2, p=1): Hout = 2*Hin. One thread per output elem.
// w layout (Cin, Cout, 4, 4).
// ---------------------------------------------------------------------------
__global__ void __launch_bounds__(256)
deconv_kernel(const float* __restrict__ in, const float* __restrict__ w,
              const float* __restrict__ bias, float* __restrict__ out,
              int Cin, int Cout, int Hin, int doRelu) {
  int Hout = Hin * 2;
  int total = Bn * Cout * Hout * Hout;
  int stride = gridDim.x * blockDim.x;
  for (int idx = blockIdx.x * blockDim.x + threadIdx.x; idx < total; idx += stride) {
    int ox = idx % Hout;
    int tmp = idx / Hout;
    int oy = tmp % Hout; tmp /= Hout;
    int oc = tmp % Cout;
    int b  = tmp / Cout;
    float acc = bias[oc];
    for (int ky = 0; ky < 4; ky++) {
      int ty = oy + 1 - ky;
      if (ty < 0 || (ty & 1)) continue;
      int iy = ty >> 1;
      if (iy >= Hin) continue;
      for (int kx = 0; kx < 4; kx++) {
        int tx = ox + 1 - kx;
        if (tx < 0 || (tx & 1)) continue;
        int ix = tx >> 1;
        if (ix >= Hin) continue;
        const float* ip = in + ((size_t)b * Cin) * Hin * Hin + iy * Hin + ix;
        const float* wp = w + (size_t)oc * 16 + ky * 4 + kx;
        for (int ic = 0; ic < Cin; ic++)
          acc += ip[(size_t)ic * Hin * Hin] * wp[(size_t)ic * Cout * 16];
      }
    }
    out[idx] = doRelu ? fmaxf(acc, 0.0f) : acc;
  }
}

// ---------------------------------------------------------------------------
// Extract [:R,:C] from (B,1,128,128), apply mask.
// ---------------------------------------------------------------------------
__global__ void logits_kernel(const float* __restrict__ h5,
                              const unsigned char* __restrict__ mask,
                              float* __restrict__ out) {
  int total = Bn * Rr * Cc;
  int stride = gridDim.x * blockDim.x;
  for (int idx = blockIdx.x * blockDim.x + threadIdx.x; idx < total; idx += stride) {
    int b  = idx / (Rr * Cc);
    int rc = idx % (Rr * Cc);
    int r = rc / Cc, c = rc % Cc;
    float v = h5[((size_t)b * 128 + r) * 128 + c];
    out[idx] = mask[idx] ? v : NEGV;
  }
}

// ---------------------------------------------------------------------------
// Host launcher
// ---------------------------------------------------------------------------
extern "C" void kernel_launch(void* const* d_in, const int* in_sizes, int n_in,
                              void* d_out, int out_size, void* d_ws, size_t ws_size,
                              hipStream_t stream) {
  (void)in_sizes; (void)n_in; (void)out_size; (void)ws_size;

  const float* node_features = (const float*)d_in[0];
  const int*   edge_index    = (const int*)d_in[1];
  const float* edge_weight   = (const float*)d_in[2];
  const int*   macro_idx     = (const int*)d_in[3];
  const float* metadata      = (const float*)d_in[4];
  const unsigned char* mask  = (const unsigned char*)d_in[5];
  const float* node_proj_w   = (const float*)d_in[6];
  const float* node_proj_b   = (const float*)d_in[7];
  const float* msg_w         = (const float*)d_in[8];
  const float* msg_b         = (const float*)d_in[9];
  const float* upd_w         = (const float*)d_in[10];
  const float* upd_b         = (const float*)d_in[11];
  const float* macro_w       = (const float*)d_in[12];
  const float* macro_b       = (const float*)d_in[13];
  const float* meta_w        = (const float*)d_in[14];
  const float* meta_b        = (const float*)d_in[15];
  const float* policy_w      = (const float*)d_in[16];
  const float* policy_b      = (const float*)d_in[17];
  const float* dc_w[5] = { (const float*)d_in[18], (const float*)d_in[20],
                           (const float*)d_in[22], (const float*)d_in[24],
                           (const float*)d_in[26] };
  const float* dc_b[5] = { (const float*)d_in[19], (const float*)d_in[21],
                           (const float*)d_in[23], (const float*)d_in[25],
                           (const float*)d_in[27] };
  const float* val_w1 = (const float*)d_in[28];
  const float* val_b1 = (const float*)d_in[29];
  const float* val_w2 = (const float*)d_in[30];
  const float* val_b2 = (const float*)d_in[31];

  float* out = (float*)d_out;

  // Workspace carving
  char* ws = (char*)d_ws;
  size_t off = 0;
  auto carve = [&](size_t bytes) -> void* {
    void* p = ws + off;
    off += (bytes + 255) & ~(size_t)255;
    return p;
  };
  _Float16* xA     = (_Float16*)carve((size_t)Bn * Nn * EMB * 2);
  _Float16* xB     = (_Float16*)carve((size_t)Bn * Nn * EMB * 2);
  float*    agg    = (float*)carve((size_t)Bn * Nn * EMB * 4);
  float*    invdeg = (float*)carve((size_t)Nn * 4);
  _Float16* pw_msg = (_Float16*)carve((size_t)Ln * 2 * 2 * 512 * 2);
  _Float16* pw_upd = (_Float16*)carve((size_t)Ln * 2 * 2 * 512 * 2);
  float*    gsum   = (float*)carve((size_t)Bn * EMB * 4);
  float*    pol    = (float*)carve((size_t)Bn * 512 * 4);
  float*    dbufA  = (float*)carve((size_t)Bn * 128 * 128 * 4);
  float*    dbufB  = (float*)carve((size_t)Bn * 128 * 128 * 4);

  // 1) pack WMMA weight fragments
  pack_weights_kernel<<<48, 256, 0, stream>>>(msg_w, upd_w, pw_msg, pw_upd);

  // 2) node projection (f32 -> f16 embeddings)
  node_proj_kernel<<<512, 256, 0, stream>>>(node_features, node_proj_w, node_proj_b, xA);

  // 3) degree -> 1/deg
  zero_f32_kernel<<<128, 256, 0, stream>>>(invdeg, (long)Nn);
  degree_kernel<<<1024, 256, 0, stream>>>(edge_index + En, invdeg);
  invdeg_kernel<<<196, 256, 0, stream>>>(invdeg);

  // 4) L message-passing layers (WMMA)
  _Float16* xcur = xA;
  _Float16* xnext = xB;
  for (int l = 0; l < Ln; l++) {
    zero_f32_kernel<<<1024, 256, 0, stream>>>(agg, (long)Bn * Nn * EMB);
    msg_kernel<<<1024, 256, 0, stream>>>(xcur, edge_index, edge_weight, invdeg,
                                         pw_msg, msg_w, msg_b, agg, l);
    upd_kernel<<<256, 256, 0, stream>>>(xcur, agg, pw_upd, upd_b, xnext, l);
    _Float16* t = xcur; xcur = xnext; xnext = t;
  }

  // 5) graph mean + heads
  gsum_kernel<<<Bn * EMB, 256, 0, stream>>>(xcur, gsum);
  head_kernel<<<1, 256, 0, stream>>>(xcur, gsum, macro_idx, metadata,
                                     macro_w, macro_b, meta_w, meta_b,
                                     policy_w, policy_b,
                                     val_w1, val_b1, val_w2, val_b2,
                                     pol, out + Bn * Rr * Cc);

  // 6) deconv stack 4x4 -> 128x128
  deconv_kernel<<<64,  256, 0, stream>>>(pol,   dc_w[0], dc_b[0], dbufA, 32, 16,  4, 1);
  deconv_kernel<<<64,  256, 0, stream>>>(dbufA, dc_w[1], dc_b[1], dbufB, 16,  8,  8, 1);
  deconv_kernel<<<64,  256, 0, stream>>>(dbufB, dc_w[2], dc_b[2], dbufA,  8,  4, 16, 1);
  deconv_kernel<<<128, 256, 0, stream>>>(dbufA, dc_w[3], dc_b[3], dbufB,  4,  2, 32, 1);
  deconv_kernel<<<256, 256, 0, stream>>>(dbufB, dc_w[4], dc_b[4], dbufA,  2,  1, 64, 0);

  // 7) masked logits
  logits_kernel<<<80, 256, 0, stream>>>(dbufA, mask, out);
}
